// SpecTraNet_24970939859573
// MI455X (gfx1250) — compile-verified, matrix operand order checked by reading, not run on hardware
//
#include <hip/hip_runtime.h>
#include <hip/hip_bf16.h>
#include <math.h>

// ---------------------------------------------------------------------------
// MI455X (gfx1250) implementation.
// All dense GEMMs run on v_wmma_f32_16x16x32_bf16 (wave32 WMMA, fp32 acc).
// Weights are pre-transposed to [N][K] bf16 so B fragments are contiguous
// 32-byte per-lane global loads in the same contiguous k order as A (k = 16*(lane/16) + s); A tiles are
// DMA'd into LDS by the Tensor Data Mover (tensor_load_to_lds, TENSORcnt,
// double buffered) so the copy overlaps the WMMAs, with a manual-staging
// fallback when the TDM builtin is unavailable.
// ---------------------------------------------------------------------------

typedef unsigned short u16;
typedef __attribute__((ext_vector_type(16))) __bf16 v16bf;
typedef __attribute__((ext_vector_type(8)))  float  v8f;
typedef __attribute__((ext_vector_type(4)))  unsigned int u32x4;
typedef __attribute__((ext_vector_type(4)))  int i32x4;
typedef __attribute__((ext_vector_type(8)))  int i32x8;

#define HAVE_TDM __has_builtin(__builtin_amdgcn_tensor_load_to_lds)

__device__ __forceinline__ u16 f2bf(float f) {
  union { float f; unsigned int u; } c; c.f = f;
  unsigned int u = c.u;
  u += 0x7fffu + ((u >> 16) & 1u);     // round-to-nearest-even
  return (u16)(u >> 16);
}
__device__ __forceinline__ float gelu_exact(float x) {
  return 0.5f * x * (1.0f + erff(x * 0.70710678118654752f));
}

#if HAVE_TDM
// Issue a TDM 2D tile load: global (bf16, row stride = stride0 elems) -> LDS.
// D# packing per CDNA5 ISA 8.3/8.4 (group0: count/lds/global/type,
// group1: data_size=2B, dims, tile, stride0). Groups 2/3 zero (<=2D tensor).
// This toolchain exposes the 6-arg builtin:
//   (u32x4 g0, i32x8 g1, i32x4 g2, i32x4 g3, i32x8 g4, i32 cpol)
__device__ __forceinline__ void tdm_load_tile(const u16* gptr, unsigned lds_off,
                                              unsigned td0, unsigned td1,
                                              unsigned tile0, unsigned tile1,
                                              unsigned stride0) {
  unsigned long long ga = (unsigned long long)(size_t)gptr;
  u32x4 g0;
  g0[0] = 1u;                                        // count=1 (user D#)
  g0[1] = lds_off;                                   // LDS byte address
  g0[2] = (unsigned)(ga & 0xffffffffu);              // global addr [31:0]
  g0[3] = (unsigned)((ga >> 32) & 0x01ffffffu)       // global addr [56:32]
          | 0x80000000u;                             // type=2 ("image")
  i32x8 g1;
  g1[0] = (int)(1u << 16);                           // data_size=1 -> 2 bytes
  g1[1] = (int)((td0 & 0xffffu) << 16);              // tensor_dim0[15:0]
  g1[2] = (int)((td0 >> 16) | ((td1 & 0xffffu) << 16));
  g1[3] = (int)((td1 >> 16) | (tile0 << 16));        // tile_dim0
  g1[4] = (int)(tile1 & 0xffffu);                    // tile_dim1 (tile_dim2=0)
  g1[5] = (int)stride0;                              // tensor_dim0_stride[31:0]
  g1[6] = 0;                                         // stride0[47:32]|stride1
  g1[7] = 0;
  i32x4 gz4 = {0, 0, 0, 0};
  i32x8 gz8 = {0, 0, 0, 0, 0, 0, 0, 0};
  __builtin_amdgcn_tensor_load_to_lds(g0, g1, gz4, gz4, gz8, 0);
}
#endif

// ------------------------------- converters --------------------------------
__global__ __launch_bounds__(256) void k_f32_to_bf16(const float* __restrict__ in,
                                                     u16* __restrict__ out, int n) {
  int i = blockIdx.x * 256 + threadIdx.x;
  if (i < n) out[i] = f2bf(in[i]);
}

// in[R][C] f32 -> out[C][R] bf16   (weight [K][N] -> Wt [N][K])
__global__ __launch_bounds__(256) void k_transpose_bf16(const float* __restrict__ in,
                                                        u16* __restrict__ out,
                                                        int R, int C) {
  int i = blockIdx.x * 256 + threadIdx.x;
  if (i >= R * C) return;
  int r = i / C, c = i % C;
  out[(size_t)c * R + r] = f2bf(in[(size_t)r * C + c]);
}

// ------------------------------- WMMA GEMM ---------------------------------
// C[M][N] (f32, +bias, optional exact GELU) = A[M][K](bf16) @ Wt[N][K](bf16)^T
// block: 256 threads = 8 waves (2 M x 4 N), block tile 64(M) x 128(N),
// wave tile 32x32 = 2x2 v_wmma_f32_16x16x32_bf16 accumulators, K step 32.
// A tile 64x32 double-buffered in LDS, filled by TDM from wave 0.
template <int ACT>
__global__ __launch_bounds__(256)
void wmma_gemm_kernel(const u16* __restrict__ A, const u16* __restrict__ Wt,
                      const float* __restrict__ bias, float* __restrict__ C,
                      int M, int N, int K) {
  __shared__ __align__(16) u16 sA[2][64 * 32];   // 2 x 4 KB A tiles
  const int lane = threadIdx.x & 31;
  const int wave = threadIdx.x >> 5;
  const int wm = wave >> 2;        // 0..1
  const int wn = wave & 3;         // 0..3
  const int mblk = blockIdx.y * 64;
  const int nblk = blockIdx.x * 128;
  const int lrow = lane & 15;
  const int khalf = (lane >> 4) * 16;

  v8f acc[2][2] = {};

#if HAVE_TDM
  const bool issuer = (wave == 0);               // wave-uniform branch
  const unsigned lds0 = (unsigned)(size_t)(void*)&sA[0][0];
  const unsigned lds1 = (unsigned)(size_t)(void*)&sA[1][0];
  const unsigned rows = (unsigned)(M - mblk);    // OOB rows zero-filled by TDM
  if (issuer) {
    tdm_load_tile(A + (size_t)mblk * K, lds0, (unsigned)K, rows, 32u, 64u,
                  (unsigned)K);
    __builtin_amdgcn_s_wait_tensorcnt(0);
  }
  __syncthreads();
  int buf = 0;
  for (int kb = 0; kb < K; kb += 32) {
    // kick off next tile into the other buffer (overlaps with WMMAs below)
    if (issuer && (kb + 32 < K))
      tdm_load_tile(A + (size_t)mblk * K + kb + 32, buf ? lds0 : lds1,
                    (unsigned)(K - kb - 32), rows, 32u, 64u, (unsigned)K);

    v16bf bfrag[2];
#pragma unroll
    for (int nj = 0; nj < 2; ++nj) {
      int colg = nblk + wn * 32 + nj * 16 + lrow;
      int colc = (colg < N) ? colg : (N - 1);
      bfrag[nj] = *(const v16bf*)(Wt + (size_t)colc * K + kb + khalf);
    }
    v16bf afrag[2];
#pragma unroll
    for (int mi = 0; mi < 2; ++mi) {
      int row = wm * 32 + mi * 16 + lrow;
      afrag[mi] = *(const v16bf*)(&sA[buf][0] + row * 32 + khalf);
    }
#pragma unroll
    for (int mi = 0; mi < 2; ++mi)
#pragma unroll
      for (int nj = 0; nj < 2; ++nj)
        acc[mi][nj] = __builtin_amdgcn_wmma_f32_16x16x32_bf16(
            false, afrag[mi], false, bfrag[nj], (short)0, acc[mi][nj],
            false, false);

    if (issuer) __builtin_amdgcn_s_wait_tensorcnt(0);  // next tile landed
    __syncthreads();
    buf ^= 1;
  }
#else
  // Fallback: manual VMEM->LDS staging (single buffer).
  for (int kb = 0; kb < K; kb += 32) {
    {
      int row = threadIdx.x >> 2;          // 0..63
      int col = (threadIdx.x & 3) * 8;     // 0,8,16,24 (u16 units)
      int grow = mblk + row;
      uint4 v = {0u, 0u, 0u, 0u};
      if (grow < M)
        v = *(const uint4*)(A + (size_t)grow * K + kb + col);
      *(uint4*)(&sA[0][0] + row * 32 + col) = v;
      if (kb + 32 < K && grow < M)
        __builtin_prefetch(A + (size_t)grow * K + kb + 32 + col, 0, 1);
    }
    __syncthreads();
    v16bf bfrag[2];
#pragma unroll
    for (int nj = 0; nj < 2; ++nj) {
      int colg = nblk + wn * 32 + nj * 16 + lrow;
      int colc = (colg < N) ? colg : (N - 1);
      bfrag[nj] = *(const v16bf*)(Wt + (size_t)colc * K + kb + khalf);
    }
    v16bf afrag[2];
#pragma unroll
    for (int mi = 0; mi < 2; ++mi) {
      int row = wm * 32 + mi * 16 + lrow;
      afrag[mi] = *(const v16bf*)(&sA[0][0] + row * 32 + khalf);
    }
#pragma unroll
    for (int mi = 0; mi < 2; ++mi)
#pragma unroll
      for (int nj = 0; nj < 2; ++nj)
        acc[mi][nj] = __builtin_amdgcn_wmma_f32_16x16x32_bf16(
            false, afrag[mi], false, bfrag[nj], (short)0, acc[mi][nj],
            false, false);
    __syncthreads();
  }
#endif

  // ---- epilogue: C/D layout = VGPR r -> rows {r, r+8}, lane%16 -> col ----
#pragma unroll
  for (int mi = 0; mi < 2; ++mi)
#pragma unroll
    for (int nj = 0; nj < 2; ++nj) {
      int colg = nblk + wn * 32 + nj * 16 + lrow;
      if (colg >= N) continue;
      float bval = bias ? bias[colg] : 0.0f;
#pragma unroll
      for (int r = 0; r < 8; ++r) {
        int rowl = (lane < 16) ? r : (r + 8);
        int rowg = mblk + wm * 32 + mi * 16 + rowl;
        if (rowg < M) {
          float v = acc[mi][nj][r] + bval;
          if (ACT == 1) v = gelu_exact(v);
          C[(size_t)rowg * N + colg] = v;
        }
      }
    }
}

// ------------------------ row L2-normalize -> bf16 -------------------------
// one block (256 thr) per row of length 256
__global__ __launch_bounds__(256)
void k_rownorm_bf16(const float* __restrict__ in, u16* __restrict__ out, int rows) {
  __shared__ float red[256];
  int row = blockIdx.x, t = threadIdx.x;
  float x = in[(size_t)row * 256 + t];
  red[t] = x * x;
  __syncthreads();
  for (int s = 128; s > 0; s >>= 1) {
    if (t < s) red[t] += red[t + s];
    __syncthreads();
  }
  float nrm = sqrtf(red[0]);
  nrm = fmaxf(nrm, 1e-12f);
  out[(size_t)row * 256 + t] = f2bf(x / nrm);
}

// ---------------- OT cost + row softmax (KP=64) + Tm stats -----------------
// one block (64 thr) per (b,n).  E = qn.pn (from WMMA GEMM).
__global__ __launch_bounds__(64)
void k_cost_softmax(const float* __restrict__ E, const float* __restrict__ coords,
                    const float* __restrict__ protoxy, float* __restrict__ Tm,
                    u16* __restrict__ Tm_bf, float* __restrict__ m1,
                    float* __restrict__ m2, int Npts) {
  __shared__ float sl[64];
  __shared__ float sb[2];
  int bn = blockIdx.x, k = threadIdx.x;
  int n = bn % Npts;
  float cy = coords[n * 2 + 0], cx = coords[n * 2 + 1];
  float py = tanhf(protoxy[k * 2 + 0]), px = tanhf(protoxy[k * 2 + 1]);
  float dy = cy - py, dx = cx - px;
  float geod = sqrtf(dy * dy + dx * dx + 1e-12f);
  float cost = geod + 1.0f - E[(size_t)bn * 64 + k];
  float logit = -(1.0f / (0.1f + 1e-6f)) * cost;   // -TAU*C/(EPS+1e-6)
  sl[k] = logit;
  __syncthreads();
  if (k == 0) {
    float mx = sl[0];
    for (int i = 1; i < 64; ++i) mx = fmaxf(mx, sl[i]);
    sb[0] = mx;
  }
  __syncthreads();
  float e = expf(logit - sb[0]);
  sl[k] = e;
  __syncthreads();
  if (k == 0) {
    float sm = 0.f;
    for (int i = 0; i < 64; ++i) sm += sl[i];
    sb[1] = sm;
  }
  __syncthreads();
  float tm = e / sb[1];
  Tm[(size_t)bn * 64 + k] = tm;
  Tm_bf[(size_t)bn * 64 + k] = f2bf(tm);
  sl[k] = tm * tm;
  __syncthreads();
  if (k == 0) {
    float s2 = 0.f;
    for (int i = 0; i < 64; ++i) s2 += sl[i];
    m1[bn] = 1.0f / 64.0f;          // softmax rows sum to 1
    m2[bn] = s2 / 64.0f;
  }
}

// ------------------------------ LayerNorm ----------------------------------
// out = LN(i1+i2)*g+b over D=256; optional bf16 copy
__global__ __launch_bounds__(256)
void k_ln(const float* __restrict__ i1, const float* __restrict__ i2,
          const float* __restrict__ g, const float* __restrict__ b,
          float* __restrict__ out, u16* __restrict__ out_bf) {
  __shared__ float red[256];
  int row = blockIdx.x, t = threadIdx.x;
  float x = i1[(size_t)row * 256 + t] + i2[(size_t)row * 256 + t];
  red[t] = x;
  __syncthreads();
  for (int s = 128; s > 0; s >>= 1) { if (t < s) red[t] += red[t + s]; __syncthreads(); }
  float mean = red[0] * (1.0f / 256.0f);
  __syncthreads();
  float d = x - mean;
  red[t] = d * d;
  __syncthreads();
  for (int s = 128; s > 0; s >>= 1) { if (t < s) red[t] += red[t + s]; __syncthreads(); }
  float var = red[0] * (1.0f / 256.0f);
  float y = d * rsqrtf(var + 1e-5f) * g[t] + b[t];
  out[(size_t)row * 256 + t] = y;
  if (out_bf) out_bf[(size_t)row * 256 + t] = f2bf(y);
}

// -------- per-point kernel generator: anisotropic metric + scale gates -----
// one block (64 thr) per (b,n)
__global__ __launch_bounds__(64)
void k_point(const float* __restrict__ h1, const float* __restrict__ m1,
             const float* __restrict__ m2, const float* __restrict__ ss_w1,
             const float* __restrict__ ss_b1, const float* __restrict__ ss_w2,
             const float* __restrict__ ss_b2, const float* __restrict__ zg_w,
             const float* __restrict__ zg_b, const float* __restrict__ ph_w,
             const float* __restrict__ ph_b, const float* __restrict__ sg_w,
             const float* __restrict__ sg_b, float* __restrict__ pA00,
             float* __restrict__ pA01, float* __restrict__ pA11,
             float* __restrict__ pg, float* __restrict__ Afro) {
  __shared__ float ker[32];
  int bn = blockIdx.x, t = threadIdx.x;
  if (t < 16) {                       // z_emb[j] = gelu(h1 . zg_w[:,j] + b)
    float acc = zg_b[t];
    const float* hrow = h1 + (size_t)bn * 256;
    for (int d = 0; d < 256; ++d) acc += hrow[d] * zg_w[d * 16 + t];
    ker[16 + t] = gelu_exact(acc);
  }
  __syncthreads();
  if (t == 0) {
    float a = m1[bn], b2 = m2[bn];
    float hid[16];
    for (int i = 0; i < 16; ++i)
      hid[i] = gelu_exact(a * ss_w1[i] + b2 * ss_w1[16 + i] + ss_b1[i]);
    for (int j = 0; j < 16; ++j) {
      float s = ss_b2[j];
      for (int i = 0; i < 16; ++i) s += hid[i] * ss_w2[i * 16 + j];
      ker[j] = gelu_exact(s);
    }
    float abc[3], sg[3];
    for (int o = 0; o < 3; ++o) {
      float pa = ph_b[o], sa = sg_b[o];
      for (int c = 0; c < 32; ++c) { pa += ker[c] * ph_w[c * 3 + o]; sa += ker[c] * sg_w[c * 3 + o]; }
      abc[o] = pa; sg[o] = sa;
    }
    float La = expf(abc[0]), Lb = abc[1], Lc = expf(abc[2]);
    float A00 = La * La, A01 = La * Lb, A11 = Lb * Lb + Lc * Lc;
    pA00[bn] = A00; pA01[bn] = A01; pA11[bn] = A11;
    float mx = fmaxf(sg[0], fmaxf(sg[1], sg[2]));
    float e0 = expf(sg[0] - mx), e1 = expf(sg[1] - mx), e2 = expf(sg[2] - mx);
    float inv = 1.0f / (e0 + e1 + e2);
    pg[bn * 3 + 0] = e0 * inv; pg[bn * 3 + 1] = e1 * inv; pg[bn * 3 + 2] = e2 * inv;
    Afro[bn] = sqrtf(A00 * A00 + 2.0f * A01 * A01 + A11 * A11 + 1e-12f);
  }
}

// ---------------- multi-scale KNN attention (KN=16) ------------------------
// one block (256 thr) per (b,n)
__global__ __launch_bounds__(256)
void k_attn(const float* __restrict__ q2, const float* __restrict__ k2,
            const float* __restrict__ v2, const float* __restrict__ Tm,
            const int* __restrict__ knn, const float* __restrict__ rel,
            const float* __restrict__ geodk, const float* __restrict__ pA00,
            const float* __restrict__ pA01, const float* __restrict__ pA11,
            const float* __restrict__ pg, float* __restrict__ attout,
            u16* __restrict__ attout_bf, int Npts) {
  __shared__ float part[256];
  __shared__ float sTm[64];
  __shared__ float cont[16], simv[16], lgs[16], att[16];
  int bn = blockIdx.x, t = threadIdx.x;
  int b = bn / Npts, n = bn % Npts;
  if (t < 64) sTm[t] = Tm[(size_t)bn * 64 + t];
  __syncthreads();
  int j = t >> 4, e = t & 15;
  int nb = knn[n * 16 + j];
  size_t nrow = ((size_t)b * Npts + nb) * 256;
  // content partials: 16 d-values per thread
  {
    float a = 0.f;
    const float* qrow = q2 + (size_t)bn * 256;
    for (int d = e * 16; d < e * 16 + 16; ++d) a += qrow[d] * k2[nrow + d];
    part[t] = a;
  }
  __syncthreads();
  if (t < 16) {
    float s = 0.f;
    for (int i = 0; i < 16; ++i) s += part[t * 16 + i];
    cont[t] = s * (1.0f / 16.0f);          // 1/sqrt(D), D=256
  }
  __syncthreads();
  // sim partials: 4 proto values per thread
  {
    const float* tnb = Tm + ((size_t)b * Npts + nb) * 64;
    float a = 0.f;
    for (int p = e * 4; p < e * 4 + 4; ++p) a += sTm[p] * tnb[p];
    part[t] = a;
  }
  __syncthreads();
  if (t < 16) {
    float s = 0.f;
    for (int i = 0; i < 16; ++i) s += part[t * 16 + i];
    simv[t] = s;
  }
  __syncthreads();
  if (t < 16) {
    const float Sr[3]  = {1.5f, 2.5f, 4.0f};
    const float Ss0[3] = {0.25f, 0.35f, 0.55f};
    const float Ssl[3] = {0.2f, 0.2f, 0.25f};
    const float Scb[3] = {0.7f, 0.5f, 0.3f};
    float geod = geodk[n * 16 + t];
    float r0 = rel[(n * 16 + t) * 2 + 0], r1 = rel[(n * 16 + t) * 2 + 1];
    float A00 = pA00[bn], A01 = pA01[bn], A11 = pA11[bn];
    float rAr = A00 * r0 * r0 + 2.0f * A01 * r0 * r1 + A11 * r1 * r1;
    float cp = expf(-geod * geod * 0.25f);           // MGAT_R0 = 2
    float lg = 0.f;
    for (int i = 0; i < 3; ++i) {
      float hm = (geod > Sr[i]) ? -10000.0f : 0.0f;
      float sg2 = Ss0[i] + Ssl[i] * geod; sg2 *= sg2;
      float li = cont[t] - rAr / fmaxf(sg2, 1e-6f) - 0.25f * geod + simv[t]
                 + Scb[i] * cp + hm;
      lg += li * pg[bn * 3 + i];
    }
    lgs[t] = lg;
  }
  __syncthreads();
  if (t < 16) {
    float mx = lgs[0];
    for (int i = 1; i < 16; ++i) mx = fmaxf(mx, lgs[i]);
    float sm = 0.f;
    for (int i = 0; i < 16; ++i) sm += expf(lgs[i] - mx);
    att[t] = expf(lgs[t] - mx) / sm;
  }
  __syncthreads();
  // weighted V gather: thread t = channel d
  {
    float o = 0.f;
    for (int jj = 0; jj < 16; ++jj) {
      int nb2 = knn[n * 16 + jj];
      o += att[jj] * v2[((size_t)b * Npts + nb2) * 256 + t];
    }
    attout[(size_t)bn * 256 + t] = o;
    attout_bf[(size_t)bn * 256 + t] = f2bf(o);
  }
}

// ------------------------------ TV regularizer -----------------------------
__global__ __launch_bounds__(256) void k_zero(float* p) { if (threadIdx.x == 0) *p = 0.f; }

__global__ __launch_bounds__(256)
void k_tv(const float* __restrict__ Afro, float* __restrict__ reg,
          int Bb, int H, int W) {
  int i = blockIdx.x * 256 + threadIdx.x;
  int tot = Bb * H * W;
  if (i >= tot) return;
  int x = i % W, y = (i / W) % H;
  float a = Afro[i], c = 0.f;
  float invV = 1.0f / (float)(Bb * (H - 1) * W);
  float invH = 1.0f / (float)(Bb * H * (W - 1));
  if (y > 0) c += fabsf(a - Afro[i - W]) * invV;
  if (x > 0) c += fabsf(a - Afro[i - 1]) * invH;
  if (c != 0.f) atomicAdd(reg, 0.001f * c);   // TV_W
}

// ===========================================================================
extern "C" void kernel_launch(void* const* d_in, const int* in_sizes, int n_in,
                              void* d_out, int out_size, void* d_ws, size_t ws_size,
                              hipStream_t stream) {
  constexpr int Bb = 4, H = 64, W = 64, D = 256, KP = 64;
  constexpr int Npts = H * W;          // 4096
  constexpr int BN = Bb * Npts;        // 16384
  constexpr int DF = 4 * D;            // 1024

  const float* z       = (const float*)d_in[0];
  const float* coords  = (const float*)d_in[1];
  const float* rel     = (const float*)d_in[2];
  const float* geodk   = (const float*)d_in[3];
  const float* proto   = (const float*)d_in[4];
  const float* protoxy = (const float*)d_in[5];
  const float* Wq_got  = (const float*)d_in[6];
  const float* Wp_got  = (const float*)d_in[7];
  const float* ln_g    = (const float*)d_in[8];
  const float* ln_b    = (const float*)d_in[9];
  const float* ffn_w1  = (const float*)d_in[10];
  const float* ffn_b1  = (const float*)d_in[11];
  const float* ffn_w2  = (const float*)d_in[12];
  const float* ffn_b2  = (const float*)d_in[13];
  const float* Wq = (const float*)d_in[14]; const float* bq = (const float*)d_in[15];
  const float* Wk = (const float*)d_in[16]; const float* bk = (const float*)d_in[17];
  const float* Wv = (const float*)d_in[18]; const float* bv = (const float*)d_in[19];
  const float* Wo = (const float*)d_in[20]; const float* bo = (const float*)d_in[21];
  const float* mln_g = (const float*)d_in[22]; const float* mln_b = (const float*)d_in[23];
  const float* ss_w1 = (const float*)d_in[24]; const float* ss_b1 = (const float*)d_in[25];
  const float* ss_w2 = (const float*)d_in[26]; const float* ss_b2 = (const float*)d_in[27];
  const float* zg_w  = (const float*)d_in[28]; const float* zg_b  = (const float*)d_in[29];
  const float* ph_w  = (const float*)d_in[30]; const float* ph_b  = (const float*)d_in[31];
  const float* sg_w  = (const float*)d_in[32]; const float* sg_b  = (const float*)d_in[33];
  const int*   knn   = (const int*)d_in[34];
  float* out = (float*)d_out;

  // ---- workspace bump allocator ----
  char* base = (char*)d_ws;
  size_t off = 0;
  auto alloc = [&](size_t bytes) -> void* {
    void* p = base + off;
    off = (off + bytes + 255) & ~(size_t)255;
    return p;
  };
  // bf16 buffers
  u16* WqgT = (u16*)alloc((size_t)D * D * 2);
  u16* WpgT = (u16*)alloc((size_t)D * D * 2);
  u16* fw1T = (u16*)alloc((size_t)DF * D * 2);
  u16* fw2T = (u16*)alloc((size_t)D * DF * 2);
  u16* WqT  = (u16*)alloc((size_t)D * D * 2);
  u16* WkT  = (u16*)alloc((size_t)D * D * 2);
  u16* WvT  = (u16*)alloc((size_t)D * D * 2);
  u16* WoT  = (u16*)alloc((size_t)D * D * 2);
  u16* z_bf    = (u16*)alloc((size_t)BN * D * 2);
  u16* proto_bf= (u16*)alloc((size_t)KP * D * 2);
  u16* qn_bf   = (u16*)alloc((size_t)BN * D * 2);
  u16* pn_bf   = (u16*)alloc((size_t)KP * D * 2);
  u16* Tm_bf   = (u16*)alloc((size_t)BN * KP * 2);
  u16* pT_bf   = (u16*)alloc((size_t)D * KP * 2);
  u16* h_bf    = (u16*)alloc((size_t)BN * D * 2);
  u16* ffn1_bf = (u16*)alloc((size_t)BN * DF * 2);
  u16* h1_bf   = (u16*)alloc((size_t)BN * D * 2);
  u16* ao_bf   = (u16*)alloc((size_t)BN * D * 2);
  // f32 buffers (q/mixed reuse slots after consumption)
  float* qbuf  = (float*)alloc((size_t)BN * D * 4);
  float* pbuf  = (float*)alloc((size_t)KP * D * 4);
  float* Ebuf  = (float*)alloc((size_t)BN * KP * 4);
  float* Tmbuf = (float*)alloc((size_t)BN * KP * 4);
  float* m1b   = (float*)alloc((size_t)BN * 4);
  float* m2b   = (float*)alloc((size_t)BN * 4);
  float* mixed = (float*)alloc((size_t)BN * D * 4);
  float* hbuf  = (float*)alloc((size_t)BN * D * 4);
  float* ffn1  = (float*)alloc((size_t)BN * DF * 4);
  float* ffn2  = mixed;                       // mixed dead after h
  float* h1buf = (float*)alloc((size_t)BN * D * 4);
  float* q2b   = (float*)alloc((size_t)BN * D * 4);
  float* k2b   = (float*)alloc((size_t)BN * D * 4);
  float* v2b   = (float*)alloc((size_t)BN * D * 4);
  float* aout  = qbuf;                        // q dead after qn
  float* projb = (float*)alloc((size_t)BN * D * 4);
  float* pA00  = (float*)alloc((size_t)BN * 4);
  float* pA01  = (float*)alloc((size_t)BN * 4);
  float* pA11  = (float*)alloc((size_t)BN * 4);
  float* pgb   = (float*)alloc((size_t)BN * 3 * 4);
  float* Afro  = (float*)alloc((size_t)BN * 4);
  (void)ws_size; (void)in_sizes; (void)n_in; (void)out_size;

  auto GD = [](int n) { return (n + 255) / 256; };
  auto gemm = [&](const u16* A, const u16* Wt, const float* bias, float* C,
                  int M, int Nn, int K, int act) {
    dim3 g((Nn + 127) / 128, (M + 63) / 64);
    if (act) wmma_gemm_kernel<1><<<g, 256, 0, stream>>>(A, Wt, bias, C, M, Nn, K);
    else     wmma_gemm_kernel<0><<<g, 256, 0, stream>>>(A, Wt, bias, C, M, Nn, K);
  };

  // 1) weight transposes + activation converts (f32 -> bf16)
  k_transpose_bf16<<<GD(D * D),  256, 0, stream>>>(Wq_got, WqgT, D, D);
  k_transpose_bf16<<<GD(D * D),  256, 0, stream>>>(Wp_got, WpgT, D, D);
  k_transpose_bf16<<<GD(D * DF), 256, 0, stream>>>(ffn_w1, fw1T, D, DF);
  k_transpose_bf16<<<GD(DF * D), 256, 0, stream>>>(ffn_w2, fw2T, DF, D);
  k_transpose_bf16<<<GD(D * D),  256, 0, stream>>>(Wq, WqT, D, D);
  k_transpose_bf16<<<GD(D * D),  256, 0, stream>>>(Wk, WkT, D, D);
  k_transpose_bf16<<<GD(D * D),  256, 0, stream>>>(Wv, WvT, D, D);
  k_transpose_bf16<<<GD(D * D),  256, 0, stream>>>(Wo, WoT, D, D);
  k_f32_to_bf16<<<GD(BN * D), 256, 0, stream>>>(z, z_bf, BN * D);
  k_f32_to_bf16<<<GD(KP * D), 256, 0, stream>>>(proto, proto_bf, KP * D);

  // 2) OT transport: q = z@Wq_got, p = proto@Wp_got, normalize, E = qn@pn^T
  gemm(z_bf, WqgT, nullptr, qbuf, BN, D, D, 0);
  gemm(proto_bf, WpgT, nullptr, pbuf, KP, D, D, 0);
  k_rownorm_bf16<<<BN, 256, 0, stream>>>(qbuf, qn_bf, BN);
  k_rownorm_bf16<<<KP, 256, 0, stream>>>(pbuf, pn_bf, KP);
  gemm(qn_bf, pn_bf, nullptr, Ebuf, BN, KP, D, 0);   // pn rows already [KP][D]
  k_cost_softmax<<<BN, 64, 0, stream>>>(Ebuf, coords, protoxy, Tmbuf, Tm_bf,
                                        m1b, m2b, Npts);
  // mixed = Tm @ p  (Wt = p^T : [D][KP])
  k_transpose_bf16<<<GD(KP * D), 256, 0, stream>>>(pbuf, pT_bf, KP, D);
  gemm(Tm_bf, pT_bf, nullptr, mixed, BN, D, KP, 0);
  k_ln<<<BN, 256, 0, stream>>>(z, mixed, ln_g, ln_b, hbuf, h_bf);

  // 3) FFN + residual LN
  gemm(h_bf, fw1T, ffn_b1, ffn1, BN, DF, D, 1);      // fused exact GELU
  k_f32_to_bf16<<<GD(BN * DF), 256, 0, stream>>>(ffn1, ffn1_bf, BN * DF);
  gemm(ffn1_bf, fw2T, ffn_b2, ffn2, BN, D, DF, 0);
  k_ln<<<BN, 256, 0, stream>>>(hbuf, ffn2, ln_g, ln_b, h1buf, h1_bf);

  // 4) q2/k2/v2 projections
  gemm(h1_bf, WqT, bq, q2b, BN, D, D, 0);
  gemm(h1_bf, WkT, bk, k2b, BN, D, D, 0);
  gemm(h1_bf, WvT, bv, v2b, BN, D, D, 0);

  // 5) per-point kernel generator + multi-scale KNN attention
  k_point<<<BN, 64, 0, stream>>>(h1buf, m1b, m2b, ss_w1, ss_b1, ss_w2, ss_b2,
                                 zg_w, zg_b, ph_w, ph_b, sg_w, sg_b,
                                 pA00, pA01, pA11, pgb, Afro);
  k_attn<<<BN, 256, 0, stream>>>(q2b, k2b, v2b, Tmbuf, knn, rel, geodk,
                                 pA00, pA01, pA11, pgb, aout, ao_bf, Npts);

  // 6) output projection + final LN -> d_out
  gemm(ao_bf, WoT, bo, projb, BN, D, D, 0);
  k_ln<<<BN, 256, 0, stream>>>(h1buf, projb, mln_g, mln_b, out, (u16*)nullptr);

  // 7) TV regularizer -> d_out[last]
  k_zero<<<1, 256, 0, stream>>>(out + (size_t)BN * D);
  k_tv<<<GD(Bb * H * W), 256, 0, stream>>>(Afro, out + (size_t)BN * D, Bb, H, W);
}
